// MultiHeadAttention_35476429865116
// MI455X (gfx1250) — matrix operand order repaired and checked by reading, not compiled, not run on hardware
//
#include <hip/hip_runtime.h>

#define DIM    768
#define NHEAD  16
#define HDIM   48
#define HPAD   64
#define BATCH  8
#define SEQ    1024
#define MTOT   (BATCH*SEQ)   // 8192

#define PITCH  40            // LDS row pitch in bf16 elems (80B: 16B-aligned, conflict-free)
#define STG    (128 * PITCH) // ushorts per LDS stage per matrix

typedef __attribute__((ext_vector_type(16))) __bf16 v16bf;
typedef __attribute__((ext_vector_type(8)))  float  v8f;

union FragBF { v16bf v; unsigned int u[8]; };

__device__ __forceinline__ unsigned short f2bf(float f) {
  unsigned int u = __float_as_uint(f);
  u += 0x7FFFu + ((u >> 16) & 1u);              // round-to-nearest-even
  return (unsigned short)(u >> 16);
}

__device__ __forceinline__ v8f wmma_bf16(const FragBF& a, const FragBF& b, v8f c) {
  return __builtin_amdgcn_wmma_f32_16x16x32_bf16(
      /*neg_a=*/false, a.v, /*neg_b=*/false, b.v,
      /*c_mod=*/(short)0, c, /*reuse_a=*/false, /*reuse_b=*/false);
}

// Async 16B global -> LDS copy (per-lane), tracked by ASYNCcnt.
__device__ __forceinline__ void async_b128(unsigned lds_off, const unsigned short* g) {
  unsigned long long ga = (unsigned long long)(size_t)g;
  asm volatile("global_load_async_to_lds_b128 %0, %1, off"
               :: "v"(lds_off), "v"(ga) : "memory");
}

// ---------------------------------------------------------------- utilities
__global__ __launch_bounds__(256) void k_cvt(const float* __restrict__ src,
                                             unsigned short* __restrict__ dst, int n) {
  for (int i = blockIdx.x * blockDim.x + threadIdx.x; i < n; i += gridDim.x * blockDim.x)
    dst[i] = f2bf(src[i]);
}

// Pack Wq|Wk|Wv transposed: dst[n][k] (n in [0,2304), k in [0,768)) = W_sel[k][n%768]
__global__ __launch_bounds__(256) void k_pack_qkv_t(const float* __restrict__ wq,
                                                    const float* __restrict__ wk,
                                                    const float* __restrict__ wv,
                                                    unsigned short* __restrict__ dst) {
  const int n = 3 * DIM * DIM;
  for (int i = blockIdx.x * blockDim.x + threadIdx.x; i < n; i += gridDim.x * blockDim.x) {
    int nn = i / DIM;
    int k  = i - nn * DIM;
    float v;
    if (nn < DIM)            v = wq[k * DIM + nn];
    else if (nn < 2 * DIM)   v = wk[k * DIM + (nn - DIM)];
    else                     v = wv[k * DIM + (nn - 2 * DIM)];
    dst[i] = f2bf(v);
  }
}

// Transpose-convert Wo: dst[n][k] = Wo[k][n]
__global__ __launch_bounds__(256) void k_cvt_t(const float* __restrict__ src,
                                               unsigned short* __restrict__ dst) {
  const int n = DIM * DIM;
  for (int i = blockIdx.x * blockDim.x + threadIdx.x; i < n; i += gridDim.x * blockDim.x) {
    int nn = i / DIM;
    int k  = i - nn * DIM;
    dst[i] = f2bf(src[k * DIM + nn]);
  }
}

__global__ __launch_bounds__(256) void k_zero4(uint4* __restrict__ p, int n) {
  uint4 z; z.x = z.y = z.z = z.w = 0u;
  for (int i = blockIdx.x * blockDim.x + threadIdx.x; i < n; i += gridDim.x * blockDim.x)
    p[i] = z;
}

// ---------------------------------------------------------------- GEMM core
// Block tile 128(M) x 128(N), K-step 32, 256 threads = 8 waves.
// Wave w owns rows [w*16, w*16+16), all 8 N-tiles of 16. Double-buffered LDS
// stages filled by global_load_async_to_lds_b128 (ASYNCcnt). Both A ([M][K])
// and Bt ([N][K], pre-transposed weights) are row-chunk copies: 128 rows x
// 32 k each = 512 x 16B chunks = 2 per thread per matrix per stage.
__device__ __forceinline__ void issue_stage(const unsigned short* __restrict__ A,
                                            const unsigned short* __restrict__ Bt,
                                            int m0, int n0, int k0,
                                            unsigned baseA, unsigned baseB, int tid) {
  int c0 = tid, c1 = tid + 256;
  int r0 = c0 >> 2, h0 = (c0 & 3) * 8;
  int r1 = c1 >> 2, h1 = (c1 & 3) * 8;
  async_b128(baseA + (unsigned)(r0 * PITCH + h0) * 2, A  + (m0 + r0) * DIM + k0 + h0);
  async_b128(baseA + (unsigned)(r1 * PITCH + h1) * 2, A  + (m0 + r1) * DIM + k0 + h1);
  async_b128(baseB + (unsigned)(r0 * PITCH + h0) * 2, Bt + (n0 + r0) * DIM + k0 + h0);
  async_b128(baseB + (unsigned)(r1 * PITCH + h1) * 2, Bt + (n0 + r1) * DIM + k0 + h1);
}

__device__ __forceinline__ void gemm_core(const unsigned short* __restrict__ A,
                                          const unsigned short* __restrict__ Bt,
                                          int m0, int n0,
                                          unsigned short* lA, unsigned short* lB,
                                          v8f c[8], int tid, int wave, int col, int half) {
  const unsigned a0 = (unsigned)(size_t)lA, a1 = a0 + STG * 2;  // byte LDS offsets
  const unsigned b0 = (unsigned)(size_t)lB, b1 = b0 + STG * 2;

  issue_stage(A, Bt, m0, n0, 0, a0, b0, tid);                   // prologue: stage 0

  for (int kt = 0; kt < DIM / 32; ++kt) {
    asm volatile("s_wait_asynccnt 0" ::: "memory");             // my stage-cur chunks landed
    __syncthreads();                                            // everyone's landed
    if (kt + 1 < DIM / 32)                                      // prefetch next stage (overlaps MMA)
      issue_stage(A, Bt, m0, n0, (kt + 1) * 32,
                  (kt & 1) ? a0 : a1, (kt & 1) ? b0 : b1, tid);

    const unsigned short* cA = lA + (kt & 1) * STG;
    const unsigned short* cB = lB + (kt & 1) * STG;
    FragBF a;
#pragma unroll
    for (int v = 0; v < 8; ++v) {
      int k = ((v >> 2) << 4) + half * 8 + ((v & 3) << 1);      // ISA A-layout
      a.u[v] = *reinterpret_cast<const unsigned int*>(&cA[(wave * 16 + col) * PITCH + k]);
    }
#pragma unroll
    for (int tn = 0; tn < 8; ++tn) {
      FragBF b;
#pragma unroll
      for (int v = 0; v < 8; ++v)                               // ISA B-layout: k = (v>>2)*16 + half*8 + (v&3)*2
        b.u[v] = *reinterpret_cast<const unsigned int*>(&cB[(tn * 16 + col) * PITCH + ((v >> 2) << 4) + half * 8 + ((v & 3) << 1)]);
      c[tn] = wmma_bf16(a, b, c[tn]);
    }
  }
}

// QKV projection: [8192,768]bf16 @ Wqkv_t[2304,768]bf16 -> scatter Q/K padded + V transposed
__global__ __launch_bounds__(256) void gemm_qkv_kernel(
    const unsigned short* __restrict__ Xb, const unsigned short* __restrict__ Wt,
    const float* __restrict__ bq, const float* __restrict__ bk, const float* __restrict__ bv,
    unsigned short* __restrict__ Qb,   // [B,H,S,64]  (q * 1/sqrt(48))
    unsigned short* __restrict__ Kb,   // [B,H,S,64]
    unsigned short* __restrict__ Vt)   // [B,H,48,S]
{
  __shared__ unsigned short lA[2 * STG];
  __shared__ unsigned short lB[2 * STG];
  const int tid = threadIdx.x, wave = tid >> 5, lane = tid & 31;
  const int col = lane & 15, half = lane >> 4;
  const int m0 = blockIdx.x * 128, n0 = blockIdx.y * 128;
  v8f c[8];
#pragma unroll
  for (int t = 0; t < 8; ++t) c[t] = (v8f){0,0,0,0,0,0,0,0};

  gemm_core(Xb, Wt, m0, n0, lA, lB, c, tid, wave, col, half);

  const float qscale = 0.14433756729740643f;  // 1/sqrt(48)
#pragma unroll
  for (int tn = 0; tn < 8; ++tn) {
    int n = n0 + tn * 16 + col;
    int which = n / DIM;
    int d = n - which * DIM;
    int h = d / HDIM;
    int hd = d - h * HDIM;
    float bias = (which == 0) ? bq[d] : (which == 1) ? bk[d] : bv[d];
#pragma unroll
    for (int j = 0; j < 8; ++j) {
      int mr = m0 + wave * 16 + half * 8 + j;   // ISA C-layout row
      int bi = mr >> 10, s = mr & 1023;
      float val = c[tn][j] + bias;
      if (which == 0)
        Qb[(((bi * NHEAD + h) << 10) + s) * HPAD + hd] = f2bf(val * qscale);
      else if (which == 1)
        Kb[(((bi * NHEAD + h) << 10) + s) * HPAD + hd] = f2bf(val);
      else
        Vt[((bi * NHEAD + h) * HDIM + hd) * SEQ + s] = f2bf(val);
    }
  }
}

// Output projection: [8192,768]bf16 @ Wo_t[768,768]bf16 + bo -> fp32
__global__ __launch_bounds__(256) void gemm_out_kernel(
    const unsigned short* __restrict__ Ab, const unsigned short* __restrict__ Wot,
    const float* __restrict__ bo, float* __restrict__ out)
{
  __shared__ unsigned short lA[2 * STG];
  __shared__ unsigned short lB[2 * STG];
  const int tid = threadIdx.x, wave = tid >> 5, lane = tid & 31;
  const int col = lane & 15, half = lane >> 4;
  const int m0 = blockIdx.x * 128, n0 = blockIdx.y * 128;
  v8f c[8];
#pragma unroll
  for (int t = 0; t < 8; ++t) c[t] = (v8f){0,0,0,0,0,0,0,0};

  gemm_core(Ab, Wot, m0, n0, lA, lB, c, tid, wave, col, half);

#pragma unroll
  for (int tn = 0; tn < 8; ++tn) {
    int n = n0 + tn * 16 + col;
    float bias = bo[n];
#pragma unroll
    for (int j = 0; j < 8; ++j) {
      int mr = m0 + wave * 16 + half * 8 + j;
      out[mr * DIM + n] = c[tn][j] + bias;
    }
  }
}

// ---------------------------------------------------------------- attention
// One wave = one 16-query tile; key blocks of 32; online softmax; O = 16x48 f32.
__global__ __launch_bounds__(256) void attn_kernel(
    const unsigned short* __restrict__ Qb, const unsigned short* __restrict__ Kb,
    const unsigned short* __restrict__ Vt, unsigned short* __restrict__ Ab)
{
  __shared__ unsigned short lP[8 * 16 * PITCH];   // wave-private 16x40 bf16 P staging
  const int tid = threadIdx.x, wave = tid >> 5, lane = tid & 31;
  const int col = lane & 15, half = lane >> 4;
  const int qblk = blockIdx.x, h = blockIdx.y, bb = blockIdx.z;
  const int s0 = qblk * 128 + wave * 16;
  const unsigned short* Qp = Qb + ((bb * NHEAD + h) << 10) * HPAD;
  const unsigned short* Kp = Kb + ((bb * NHEAD + h) << 10) * HPAD;
  const unsigned short* Vp = Vt + (bb * NHEAD + h) * HDIM * SEQ;
  unsigned short* myP = &lP[wave * 16 * PITCH];

  FragBF q0, q1;
#pragma unroll
  for (int v = 0; v < 8; ++v) {
    int k = ((v >> 2) << 4) + half * 8 + ((v & 3) << 1);
    q0.u[v] = *reinterpret_cast<const unsigned int*>(&Qp[(s0 + col) * HPAD + k]);
    q1.u[v] = *reinterpret_cast<const unsigned int*>(&Qp[(s0 + col) * HPAD + 32 + k]);
  }

  float mrow[8], lrow[8];
#pragma unroll
  for (int j = 0; j < 8; ++j) { mrow[j] = -3.0e38f; lrow[j] = 0.0f; }
  v8f o0 = (v8f){0,0,0,0,0,0,0,0}, o1 = o0, o2 = o0;

  for (int key0 = 0; key0 < SEQ; key0 += 32) {
    FragBF ka0, ka1, kb0, kb1;
    const unsigned short* KrA = Kp + (key0 + col) * HPAD;
    const unsigned short* KrB = KrA + 16 * HPAD;
#pragma unroll
    for (int v = 0; v < 8; ++v) {
      int kk = ((v >> 2) << 4) + half * 8 + ((v & 3) << 1);
      ka0.u[v] = *reinterpret_cast<const unsigned int*>(&KrA[kk]);
      ka1.u[v] = *reinterpret_cast<const unsigned int*>(&KrA[32 + kk]);
      kb0.u[v] = *reinterpret_cast<const unsigned int*>(&KrB[kk]);
      kb1.u[v] = *reinterpret_cast<const unsigned int*>(&KrB[32 + kk]);
    }
    v8f sA = (v8f){0,0,0,0,0,0,0,0}, sB = sA;
    sA = wmma_bf16(q0, ka0, sA); sA = wmma_bf16(q1, ka1, sA);
    sB = wmma_bf16(q0, kb0, sB); sB = wmma_bf16(q1, kb1, sB);

#pragma unroll
    for (int j = 0; j < 8; ++j) {
      float mj = fmaxf(sA[j], sB[j]);
      mj = fmaxf(mj, __shfl_xor(mj, 1));
      mj = fmaxf(mj, __shfl_xor(mj, 2));
      mj = fmaxf(mj, __shfl_xor(mj, 4));
      mj = fmaxf(mj, __shfl_xor(mj, 8));     // uniform within each 16-lane half
      float mnew  = fmaxf(mrow[j], mj);
      float alpha = __expf(mrow[j] - mnew);
      float pa = __expf(sA[j] - mnew);
      float pb = __expf(sB[j] - mnew);
      float rs = pa + pb;
      rs += __shfl_xor(rs, 1);
      rs += __shfl_xor(rs, 2);
      rs += __shfl_xor(rs, 4);
      rs += __shfl_xor(rs, 8);
      lrow[j] = lrow[j] * alpha + rs;
      mrow[j] = mnew;
      o0[j] *= alpha; o1[j] *= alpha; o2[j] *= alpha;
      int pr = (half * 8 + j) * PITCH;        // C-layout -> row-major P in LDS
      myP[pr + col]      = f2bf(pa);
      myP[pr + col + 16] = f2bf(pb);
    }
    asm volatile("s_wait_dscnt 0" ::: "memory");  // DS in-order per wave; fence compiler too
    FragBF pf;
#pragma unroll
    for (int v = 0; v < 8; ++v) {
      int k = ((v >> 2) << 4) + half * 8 + ((v & 3) << 1);
      pf.u[v] = *reinterpret_cast<const unsigned int*>(&myP[col * PITCH + k]);
    }
#pragma unroll
    for (int t = 0; t < 3; ++t) {
      FragBF vf;
      const unsigned short* Vr = Vp + (t * 16 + col) * SEQ + key0;
#pragma unroll
      for (int v = 0; v < 8; ++v)
        vf.u[v] = *reinterpret_cast<const unsigned int*>(&Vr[((v >> 2) << 4) + half * 8 + ((v & 3) << 1)]);
      if (t == 0)      o0 = wmma_bf16(pf, vf, o0);
      else if (t == 1) o1 = wmma_bf16(pf, vf, o1);
      else             o2 = wmma_bf16(pf, vf, o2);
    }
  }

#pragma unroll
  for (int j = 0; j < 8; ++j) {
    float inv = 1.0f / lrow[j];
    int m = (bb << 10) + s0 + half * 8 + j;
    unsigned short* op = Ab + m * DIM + h * HDIM + col;
    op[0]  = f2bf(o0[j] * inv);
    op[16] = f2bf(o1[j] * inv);
    op[32] = f2bf(o2[j] * inv);
  }
}

// ---------------------------------------------------------------- launcher
extern "C" void kernel_launch(void* const* d_in, const int* in_sizes, int n_in,
                              void* d_out, int out_size, void* d_ws, size_t ws_size,
                              hipStream_t stream) {
  (void)in_sizes; (void)n_in; (void)out_size; (void)ws_size;
  const float* x  = (const float*)d_in[0];
  const float* Wq = (const float*)d_in[1];
  const float* bq = (const float*)d_in[2];
  const float* Wk = (const float*)d_in[3];
  const float* bk = (const float*)d_in[4];
  const float* Wv = (const float*)d_in[5];
  const float* bv = (const float*)d_in[6];
  const float* Wo = (const float*)d_in[7];
  const float* bo = (const float*)d_in[8];
  float* out = (float*)d_out;

  char* ws = (char*)d_ws;
  unsigned short* Xb   = (unsigned short*)(ws);               // 8192*768   bf16 = 12582912 B
  unsigned short* Wqkv = (unsigned short*)(ws + 12582912);    // 2304*768   bf16 (transposed) = 3538944 B
  unsigned short* Wob  = (unsigned short*)(ws + 16121856);    // 768*768    bf16 (transposed) = 1179648 B
  unsigned short* Qb   = (unsigned short*)(ws + 17301504);    // 128*1024*64 bf16 = 16777216 B
  unsigned short* Kb   = (unsigned short*)(ws + 34078720);    // 16777216 B (contiguous after Qb)
  unsigned short* Vt   = (unsigned short*)(ws + 50855936);    // 128*48*1024 bf16 = 12582912 B
  unsigned short* Ab   = (unsigned short*)(ws + 63438848);    // 8192*768   bf16 = 12582912 B
  // total = 76021760 B

  k_cvt       <<<2048, 256, 0, stream>>>(x,  Xb,  MTOT * DIM);
  k_pack_qkv_t<<<2048, 256, 0, stream>>>(Wq, Wk, Wv, Wqkv);
  k_cvt_t     <<<1024, 256, 0, stream>>>(Wo, Wob);
  k_zero4     <<<2048, 256, 0, stream>>>((uint4*)Qb, (2 * 16777216) / 16);  // zero Qb+Kb (padding)

  gemm_qkv_kernel<<<dim3(MTOT / 128, (3 * DIM) / 128), 256, 0, stream>>>(
      Xb, Wqkv, bq, bk, bv, Qb, Kb, Vt);

  attn_kernel<<<dim3(SEQ / 128, NHEAD, BATCH), 256, 0, stream>>>(Qb, Kb, Vt, Ab);

  gemm_out_kernel<<<dim3(MTOT / 128, DIM / 128), 256, 0, stream>>>(Ab, Wob, bo, out);
}
